// GRU_29334626632382
// MI455X (gfx1250) — matrix operand order repaired and checked by reading, not compiled, not run on hardware
//
#include <hip/hip_runtime.h>
#include <stdint.h>

typedef unsigned short u16;

constexpr int HD   = 512;     // hidden size
constexpr int BZ   = 32;      // batch
constexpr int SEQ  = 2048;    // sequence length
constexpr int G3   = 1536;    // 3*HD
constexpr int KD   = 512;     // contraction dim (== HD for both layers)

typedef __attribute__((ext_vector_type(16))) __bf16 v16bf;
typedef __attribute__((ext_vector_type(8)))  float  v8f;
typedef __attribute__((ext_vector_type(4)))  int    v4i;

union Frag { v16bf v; float4 q[2]; };

// ---------------- async global->LDS helpers (guarded probe) ----------------
#if defined(__has_builtin)
#if __has_builtin(__builtin_amdgcn_global_load_async_to_lds_b128)
#define HAVE_ASYNC_LDS 1
#endif
#endif

__device__ __forceinline__ void copy16_g2l(const void* g, void* l) {
#ifdef HAVE_ASYNC_LDS
  __builtin_amdgcn_global_load_async_to_lds_b128(
      (__attribute__((address_space(1))) v4i*)(g),
      (__attribute__((address_space(3))) v4i*)(l), 0, 0);
#else
  *(float4*)l = *(const float4*)g;
#endif
}

__device__ __forceinline__ void wait_async0() {
#ifdef HAVE_ASYNC_LDS
#if __has_builtin(__builtin_amdgcn_s_wait_asynccnt)
  __builtin_amdgcn_s_wait_asynccnt(0);
#else
  asm volatile("s_wait_asynccnt 0x0" ::: "memory");
#endif
#endif
}

__device__ __forceinline__ u16 f2bf(float x) {
  unsigned u = __float_as_uint(x);
  unsigned r = (u + 0x7FFFu + ((u >> 16) & 1u)) >> 16;
  return (u16)r;
}

// ---------------- fp32 -> bf16 conversion (vectorized) ----------------
__global__ __launch_bounds__(256) void cvt4(const float4* __restrict__ in,
                                            uint2* __restrict__ out, int n4) {
  int i = blockIdx.x * blockDim.x + threadIdx.x;
  int stride = gridDim.x * blockDim.x;
  for (; i < n4; i += stride) {
    float4 f = in[i];
    uint2 o;
    o.x = (unsigned)f2bf(f.x) | ((unsigned)f2bf(f.y) << 16);
    o.y = (unsigned)f2bf(f.z) | ((unsigned)f2bf(f.w) << 16);
    out[i] = o;
  }
}

__global__ __launch_bounds__(256) void init_state(float* hf0, u16* hb0, unsigned* bar) {
  int i = blockIdx.x * blockDim.x + threadIdx.x;
  if (i < BZ * HD) { hf0[i] = 0.f; hb0[i] = 0; }
  if (i < 2) bar[i] = 0;
}

// ---------------- input-projection GEMM: C[m][n] = A[m][:] . B[n][:] + bias[n]
// A: [M][512] bf16 row-major, B: [1536][512] bf16 row-major (== W, giving x@W^T)
constexpr int TM = 128, TN = 128, TK = 32;
constexpr int ASTR = 40;  // bf16 stride (pad 32->40, keeps 16B alignment)

__global__ __launch_bounds__(256) void gemm_bf16(const u16* __restrict__ A,
                                                 const u16* __restrict__ Bm,
                                                 const float* __restrict__ bias,
                                                 float* __restrict__ C) {
  __shared__ u16 As[TM * ASTR];
  __shared__ u16 Bs[TN * ASTR];
  const int t = threadIdx.x;
  const int w = t >> 5, lane = t & 31;
  const int l16 = lane & 15, lh = lane >> 4;
  const int bm = blockIdx.y, bn = blockIdx.x;
  const int wm = w & 1, wn = w >> 1;  // wave tile: 64(M) x 32(N)

  v8f acc[4][2];
#pragma unroll
  for (int fm = 0; fm < 4; ++fm)
#pragma unroll
    for (int fn = 0; fn < 2; ++fn)
      acc[fm][fn] = (v8f){0.f, 0.f, 0.f, 0.f, 0.f, 0.f, 0.f, 0.f};

  for (int kk = 0; kk < KD; kk += TK) {
    __syncthreads();
    // stage 128x32 bf16 tiles of A and B (512 16B chunks each, 2 per thread)
#pragma unroll
    for (int i = 0; i < 2; ++i) {
      int c = t + i * 256;
      int r = c >> 2, kg = c & 3;  // row, 8-elem K group
      copy16_g2l(A + ((size_t)(bm * TM + r) * KD + kk + kg * 8), As + r * ASTR + kg * 8);
      copy16_g2l(Bm + ((size_t)(bn * TN + r) * KD + kk + kg * 8), Bs + r * ASTR + kg * 8);
    }
    wait_async0();
    __syncthreads();

    Frag a[4], b[2];
#pragma unroll
    for (int fm = 0; fm < 4; ++fm) {
      int row = wm * 64 + fm * 16 + l16;
      int kb = lh * 16;  // A lane layout: 16 contiguous K
      a[fm].q[0] = *(const float4*)&As[row * ASTR + kb];
      a[fm].q[1] = *(const float4*)&As[row * ASTR + kb + 8];
    }
#pragma unroll
    for (int fn = 0; fn < 2; ++fn) {
      int row = wn * 32 + fn * 16 + l16;
      int kb = lh * 16;  // B lane layout: 16 contiguous K
      b[fn].q[0] = *(const float4*)&Bs[row * ASTR + kb];
      b[fn].q[1] = *(const float4*)&Bs[row * ASTR + kb + 8];
    }
#pragma unroll
    for (int fm = 0; fm < 4; ++fm)
#pragma unroll
      for (int fn = 0; fn < 2; ++fn)
        acc[fm][fn] = __builtin_amdgcn_wmma_f32_16x16x32_bf16(
            false, a[fm].v, false, b[fn].v, (short)0, acc[fm][fn], false, false);
  }

#pragma unroll
  for (int fm = 0; fm < 4; ++fm)
#pragma unroll
    for (int fn = 0; fn < 2; ++fn) {
      int n = bn * TN + wn * 32 + fn * 16 + l16;
      float bv = bias[n];
      int mbase = bm * TM + wm * 64 + fm * 16 + lh * 8;
#pragma unroll
      for (int v = 0; v < 8; ++v)
        C[(size_t)(mbase + v) * G3 + n] = acc[fm][fn][v] + bv;
    }
}

// ---------------- persistent GRU recurrence ----------------
// 8 workgroups; wg owns 64 H-columns; its 192 rows of Wh (r/z/n slices) live
// in LDS for the whole sequence.  h (32x512 bf16) is double-buffered in global
// (L2) and re-staged into LDS each step; one grid barrier per step.
constexpr int RNWG = 8;
constexpr int JSL = 64;      // H columns per wg
constexpr int WSTR = 520;    // bf16 stride for K=512 rows (pad, 16B-aligned)

__global__ __launch_bounds__(256) void gru_recurrent(
    const float* __restrict__ ip, const u16* __restrict__ Whb,
    const float* __restrict__ bh, float* __restrict__ hf0, float* __restrict__ hf1,
    u16* __restrict__ hb0, u16* __restrict__ hb1, u16* __restrict__ ybf,
    float* __restrict__ yf, float* __restrict__ hlast, unsigned* __restrict__ bar) {
  extern __shared__ char smem[];
  u16* Ws = (u16*)smem;                                 // [3*JSL][WSTR]
  u16* Hsm = (u16*)(smem + (size_t)3 * JSL * WSTR * 2); // [BZ][WSTR]

  const int t = threadIdx.x;
  const int w = t >> 5, lane = t & 31;
  const int l16 = lane & 15, lh = lane >> 4;
  const int mf = w & 1, nf = w >> 1;  // wave -> (M half, 16-col group)
  const int j0 = blockIdx.x * JSL;
  const int jc = j0 + nf * 16 + l16;  // this lane's H column

  // one-time: Wh slice -> LDS (192 rows x 64 chunks of 16B)
  for (int c = t; c < 3 * JSL * 64; c += 256) {
    int r = c >> 6, kg = c & 63;
    int g = r >> 6, i = r & 63;  // gate, local row
    copy16_g2l(Whb + ((size_t)(g * HD + j0 + i) * KD + kg * 8), Ws + r * WSTR + kg * 8);
  }
  const float bhr = bh[jc], bhz = bh[HD + jc], bhn = bh[2 * HD + jc];
  wait_async0();
  __syncthreads();

  unsigned gen_target = 0;
  for (int s = 0; s < SEQ; ++s) {
    const u16* hbr = (s & 1) ? hb1 : hb0;
    u16* hbw = (s & 1) ? hb0 : hb1;
    const float* hfr = (s & 1) ? hf1 : hf0;
    float* hfw = (s & 1) ? hf0 : hf1;

    // stage h (32x512 bf16) into LDS
    for (int c = t; c < BZ * 64; c += 256) {
      int r = c >> 6, kg = c & 63;
      copy16_g2l(hbr + ((size_t)r * HD + kg * 8), Hsm + r * WSTR + kg * 8);
    }
    wait_async0();
    __syncthreads();

    v8f ar = (v8f){0.f, 0.f, 0.f, 0.f, 0.f, 0.f, 0.f, 0.f};
    v8f az = ar, an = ar;
    for (int kk = 0; kk < KD; kk += 32) {
      Frag a, br, bz, bn;
      int arow = mf * 16 + l16;
      int kb = kk + lh * 16;
      a.q[0] = *(const float4*)&Hsm[arow * WSTR + kb];
      a.q[1] = *(const float4*)&Hsm[arow * WSTR + kb + 8];
      int brow = nf * 16 + l16;
      int kbb = kk + lh * 16;
      br.q[0] = *(const float4*)&Ws[(0 * JSL + brow) * WSTR + kbb];
      br.q[1] = *(const float4*)&Ws[(0 * JSL + brow) * WSTR + kbb + 8];
      bz.q[0] = *(const float4*)&Ws[(1 * JSL + brow) * WSTR + kbb];
      bz.q[1] = *(const float4*)&Ws[(1 * JSL + brow) * WSTR + kbb + 8];
      bn.q[0] = *(const float4*)&Ws[(2 * JSL + brow) * WSTR + kbb];
      bn.q[1] = *(const float4*)&Ws[(2 * JSL + brow) * WSTR + kbb + 8];
      ar = __builtin_amdgcn_wmma_f32_16x16x32_bf16(false, a.v, false, br.v, (short)0, ar, false, false);
      az = __builtin_amdgcn_wmma_f32_16x16x32_bf16(false, a.v, false, bz.v, (short)0, az, false, false);
      an = __builtin_amdgcn_wmma_f32_16x16x32_bf16(false, a.v, false, bn.v, (short)0, an, false, false);
    }

    // gates + state update for the 8 batch rows this lane owns
#pragma unroll
    for (int v = 0; v < 8; ++v) {
      int b_ = mf * 16 + lh * 8 + v;
      size_t mrow = (size_t)b_ * SEQ + s;
      float ipr = ip[mrow * G3 + jc];
      float ipz = ip[mrow * G3 + HD + jc];
      float ipn = ip[mrow * G3 + 2 * HD + jc];
      float hpr = ar[v] + bhr, hpz = az[v] + bhz, hpn = an[v] + bhn;
      float r = 1.f / (1.f + __expf(-(ipr + hpr)));
      float z = 1.f / (1.f + __expf(-(ipz + hpz)));
      float u = ipn + r * hpn;
      float e = __expf(-2.f * u);
      float nn = (1.f - e) / (1.f + e);  // tanh
      float hold = hfr[b_ * HD + jc];
      float hnew = (1.f - z) * nn + z * hold;
      hfw[b_ * HD + jc] = hnew;
      hbw[b_ * HD + jc] = f2bf(hnew);
      if (ybf) ybf[mrow * HD + jc] = f2bf(hnew);
      if (yf) yf[mrow * HD + jc] = hnew;
      if (hlast && s == SEQ - 1) hlast[b_ * HD + jc] = hnew;
    }

    // grid barrier (all step-s writes visible before any step-s+1 read)
    __syncthreads();
    ++gen_target;
    if (t == 0) {
      __threadfence();
      unsigned prev = atomicAdd(&bar[0], 1u);
      if (prev == RNWG - 1) {
        atomicExch(&bar[0], 0u);
        __threadfence();
        atomicAdd(&bar[1], 1u);
      } else {
        while (atomicAdd(&bar[1], 0u) < gen_target) __builtin_amdgcn_s_sleep(2);
      }
    }
    __syncthreads();
    __threadfence();
  }
}

// ---------------- orchestration ----------------
extern "C" void kernel_launch(void* const* d_in, const int* in_sizes, int n_in,
                              void* d_out, int out_size, void* d_ws, size_t ws_size,
                              hipStream_t stream) {
  const float* x = (const float*)d_in[0];
  const float* Wi0 = (const float*)d_in[1];
  const float* bi0 = (const float*)d_in[2];
  const float* Wh0 = (const float*)d_in[3];
  const float* bh0 = (const float*)d_in[4];
  const float* Wi1 = (const float*)d_in[5];
  const float* bi1 = (const float*)d_in[6];
  const float* Wh1 = (const float*)d_in[7];
  const float* bh1 = (const float*)d_in[8];
  float* y1 = (float*)d_out;
  float* h1 = y1 + (size_t)BZ * SEQ * HD;

  char* ws = (char*)d_ws;
  size_t off = 0;
  auto alloc = [&](size_t bytes) {
    void* p = ws + off;
    off = (off + bytes + 255) & ~(size_t)255;
    return p;
  };
  u16* xb = (u16*)alloc((size_t)BZ * SEQ * KD * 2);
  u16* y0b = (u16*)alloc((size_t)BZ * SEQ * HD * 2);
  u16* Wi0b = (u16*)alloc((size_t)G3 * KD * 2);
  u16* Wh0b = (u16*)alloc((size_t)G3 * KD * 2);
  u16* Wi1b = (u16*)alloc((size_t)G3 * KD * 2);
  u16* Wh1b = (u16*)alloc((size_t)G3 * KD * 2);
  float* ip = (float*)alloc((size_t)BZ * SEQ * G3 * 4);
  float* hfA = (float*)alloc((size_t)BZ * HD * 4);
  float* hfB = (float*)alloc((size_t)BZ * HD * 4);
  u16* hbA = (u16*)alloc((size_t)BZ * HD * 2);
  u16* hbB = (u16*)alloc((size_t)BZ * HD * 2);
  unsigned* bar = (unsigned*)alloc(256);

  // bf16 copies of x and weights
  cvt4<<<4096, 256, 0, stream>>>((const float4*)x, (uint2*)xb, BZ * SEQ * KD / 4);
  cvt4<<<512, 256, 0, stream>>>((const float4*)Wi0, (uint2*)Wi0b, G3 * KD / 4);
  cvt4<<<512, 256, 0, stream>>>((const float4*)Wh0, (uint2*)Wh0b, G3 * KD / 4);
  cvt4<<<512, 256, 0, stream>>>((const float4*)Wi1, (uint2*)Wi1b, G3 * KD / 4);
  cvt4<<<512, 256, 0, stream>>>((const float4*)Wh1, (uint2*)Wh1b, G3 * KD / 4);

  dim3 ggrid(G3 / TN, BZ * SEQ / TM);
  size_t shmem = (size_t)(3 * JSL + BZ) * WSTR * 2;

  // layer 0
  init_state<<<64, 256, 0, stream>>>(hfA, hbA, bar);
  gemm_bf16<<<ggrid, 256, 0, stream>>>(xb, Wi0b, bi0, ip);
  gru_recurrent<<<RNWG, 256, shmem, stream>>>(ip, Wh0b, bh0, hfA, hfB, hbA, hbB,
                                              y0b, nullptr, nullptr, bar);
  // layer 1
  init_state<<<64, 256, 0, stream>>>(hfA, hbA, bar);
  gemm_bf16<<<ggrid, 256, 0, stream>>>(y0b, Wi1b, bi1, ip);
  gru_recurrent<<<RNWG, 256, shmem, stream>>>(ip, Wh1b, bh1, hfA, hfB, hbA, hbB,
                                              nullptr, y1, h1, bar);
}
